// MessagePassing_50525995270812
// MI455X (gfx1250) — compile-verified, matrix operand order checked by reading, not run on hardware
//
#include <hip/hip_runtime.h>

// ---------------------------------------------------------------------------
// Equivariant GNN forward (3 message-passing layers), MI455X / gfx1250.
// WMMA f32_16x16x32_f16 everywhere (f16 mul, f32 acc); fused, persistent
// edge MLP (fc1 -> silu -> fc2 -> messages -> atomic scatter): weights are
// staged to LDS once per block, edge tiles grid-strided, all E-sized
// intermediates live in LDS. Fragments come from b128 loads.
// ---------------------------------------------------------------------------

typedef __attribute__((ext_vector_type(16))) _Float16 v16h;
typedef __attribute__((ext_vector_type(8)))  _Float16 v8h;
typedef __attribute__((ext_vector_type(4)))  _Float16 v4h;
typedef __attribute__((ext_vector_type(8)))  float    v8f;

#define N_NODES 10000
#define N_EDGES 160000
#define INV_SQRT3 0.5773502691896258f
#define EDGE_BLOCKS 2500

__device__ __forceinline__ float sigmoid_f(float x) {
    return __builtin_amdgcn_rcpf(1.0f + __expf(-x));   // v_exp + v_rcp only
}
__device__ __forceinline__ float silu_f(float x) { return x * sigmoid_f(x); }

// Build a 16x32-f16 A/B fragment from global f32 (two 8-wide contiguous runs).
// lanes 0-15: K = {0..7, 16..23}; lanes 16-31: K = {8..15, 24..31}.
__device__ __forceinline__ v16h frag_from_global(const float* __restrict__ p,
                                                 int kb, float rs) {
    const float4 a0 = *(const float4*)(p + kb);
    const float4 a1 = *(const float4*)(p + kb + 4);
    const float4 a2 = *(const float4*)(p + kb + 16);
    const float4 a3 = *(const float4*)(p + kb + 20);
    v16h a;
    a[0]  = (_Float16)(a0.x * rs); a[1]  = (_Float16)(a0.y * rs);
    a[2]  = (_Float16)(a0.z * rs); a[3]  = (_Float16)(a0.w * rs);
    a[4]  = (_Float16)(a1.x * rs); a[5]  = (_Float16)(a1.y * rs);
    a[6]  = (_Float16)(a1.z * rs); a[7]  = (_Float16)(a1.w * rs);
    a[8]  = (_Float16)(a2.x * rs); a[9]  = (_Float16)(a2.y * rs);
    a[10] = (_Float16)(a2.z * rs); a[11] = (_Float16)(a2.w * rs);
    a[12] = (_Float16)(a3.x * rs); a[13] = (_Float16)(a3.y * rs);
    a[14] = (_Float16)(a3.z * rs); a[15] = (_Float16)(a3.w * rs);
    return a;
}

// Fragment from f16 LDS row: two ds_load_b128 per fragment.
__device__ __forceinline__ v16h frag_from_lds(const _Float16* base, int kb) {
    const v8h lo = *(const v8h*)(base + kb);
    const v8h hi = *(const v8h*)(base + kb + 16);
    v16h r;
#pragma unroll
    for (int j = 0; j < 8; ++j) { r[j] = lo[j]; r[8 + j] = hi[j]; }
    return r;
}

// ---------------------------------------------------------------------------
// Generic GEMM: C[M,Nc] = act( (A[M,K]*rowscale) @ B[K,Nc] * scale [+ resid] )
// Block = 128 threads = 4 waves; B^T tile (K x 16, f16) staged in LDS and
// shared by all 4 waves; each wave owns one 16x16 output tile.
// ---------------------------------------------------------------------------
template<int ACT, bool RESID>
__global__ __launch_bounds__(128)
void gemm_wmma_f16(const float* __restrict__ A, const float* __restrict__ rowscale,
                   const float* __restrict__ B, const float* __restrict__ resid,
                   float* __restrict__ C, int M, int K, int Nc, float scale)
{
    __shared__ _Float16 Bt[16 * 96];            // [col][k], K <= 96
    const int t = threadIdx.x;
    const int lane = t & 31;
    const int wave = t >> 5;
    const int n0 = blockIdx.x * 16;

    // stage B^T as f16 (each task: 4 consecutive k for one column)
    for (int i = t; i < 4 * K; i += 128) {
        const int c = i & 15, kc = (i >> 4) * 4;
        v4h pk;
#pragma unroll
        for (int q = 0; q < 4; ++q)
            pk[q] = (_Float16)B[(size_t)(kc + q) * Nc + n0 + c];
        *(v4h*)&Bt[c * K + kc] = pk;
    }
    __syncthreads();

    const int m0 = (blockIdx.y * 4 + wave) * 16;
    if (m0 >= M) return;                         // no barriers after this point

    const int row = lane & 15, col = lane & 15;
    const int kb = (lane & 16) ? 8 : 0;
    const float* Arow = A + (size_t)(m0 + row) * K;
    const float rs = rowscale[m0 + row];
    const _Float16* Bcol = Bt + col * K;

    v8f acc = {};
    for (int k0 = 0; k0 < K; k0 += 32) {
        const v16h a = frag_from_global(Arow + k0, kb, rs);
        const v16h b = frag_from_lds(Bcol + k0, kb);
        acc = __builtin_amdgcn_wmma_f32_16x16x32_f16(
            false, a, false, b, (short)0, acc, false, false);
    }

    const size_t base = (size_t)(m0 + ((lane & 16) ? 8 : 0)) * Nc + n0 + col;
    float* Cp = C + base;
    const float* Rp = resid + base;
#pragma unroll
    for (int r = 0; r < 8; ++r) {
        float x = acc[r] * scale;
        if (RESID) x += Rp[(size_t)r * Nc];
        if (ACT == 1) x = silu_f(x);
        Cp[(size_t)r * Nc] = x;
    }
}

// ---------------------------------------------------------------------------
// Fused, persistent per-edge kernel. One tile = 16 edges. Per block:
//   stage W1^T / W2^T (f16) once, then grid-stride over tiles:
//     h = silu(es @ W1 / 8)   (WMMA; h in LDS as f16)
//     w = h @ W2 / 8          (WMMA; w in LDS as f32)
//     messages m0..m3 from gathered xs[src]/xv[src] and edge_attr,
//     scattered with global f32 atomics (1/sqrt(16) folded in).
// LAYER: 1 -> NOUT=128 (m_s,m_v); 2 -> 192 (m0..m3); 3 -> 96 (m0,m3).
// ---------------------------------------------------------------------------
template<int LAYER>
__global__ __launch_bounds__(128)
void edge_fused(const float* __restrict__ es,     // E x 64
                const float* __restrict__ eattr,  // E x 4
                const int*   __restrict__ esrc,
                const int*   __restrict__ edst,
                const float* __restrict__ W1,     // 64 x 64
                const float* __restrict__ W2,     // 64 x NOUT
                const float* __restrict__ xs,     // N x 64
                const float* __restrict__ xv,     // 3 planes N x 32
                size_t pvs_in,
                float* __restrict__ outS, int ldS,
                float* __restrict__ outV, int ldV, size_t pvs_out,
                int ntiles)
{
    constexpr int NOUT = (LAYER == 1) ? 128 : (LAYER == 2) ? 192 : 96;
    __shared__ _Float16 es_lds[16 * 64];
    __shared__ _Float16 h_lds [16 * 64];
    __shared__ _Float16 W1t   [64 * 64];           // [n][k] f16
    __shared__ _Float16 W2t   [NOUT * 64];         // [n][k] f16
    __shared__ float    w_lds [16 * NOUT];
    __shared__ float    xs_lds[16 * 64];
    __shared__ float    xv_lds[(LAYER >= 2) ? 16 * 96 : 4];
    __shared__ float    ev_lds[16 * 4];
    __shared__ int      src_lds[16];
    __shared__ int      dst_lds[16];

    const int t    = threadIdx.x;
    const int lane = t & 31;
    const int wave = t >> 5;
    const int row = lane & 15, col = lane & 15;
    const int kb = (lane & 16) ? 8 : 0;
    const int rb = (lane & 16) ? 8 : 0;

    // ---- stage weights once per (persistent) block
    for (int i = t; i < 64 * 16; i += 128) {       // W1^T
        const int c = i >> 4, kc = (i & 15) * 4;
        v4h pk;
#pragma unroll
        for (int q = 0; q < 4; ++q)
            pk[q] = (_Float16)W1[(size_t)(kc + q) * 64 + c];
        *(v4h*)&W1t[c * 64 + kc] = pk;
    }
    for (int i = t; i < NOUT * 16; i += 128) {     // W2^T
        const int c = i >> 4, kc = (i & 15) * 4;
        v4h pk;
#pragma unroll
        for (int q = 0; q < 4; ++q)
            pk[q] = (_Float16)W2[(size_t)(kc + q) * NOUT + c];
        *(v4h*)&W2t[c * 64 + kc] = pk;
    }

    for (int tile = blockIdx.x; tile < ntiles; tile += gridDim.x) {
        const int e0 = tile * 16;

        if (t < 16) {
            src_lds[t] = esrc[e0 + t];
            dst_lds[t] = edst[e0 + t];
            *(float4*)&ev_lds[t * 4] = *(const float4*)(eattr + (size_t)(e0 + t) * 4);
        }
        __syncthreads();   // (also covers weight staging on first iteration)

        // stage edge_scalars as f16 + gather source-node features (b128)
        for (int i = t; i < 16 * 16; i += 128) {   // 256 float4 tasks each
            const int e = i >> 4, c4 = (i & 15) * 4;
            const float4 p = *(const float4*)(es + (size_t)(e0 + e) * 64 + c4);
            v4h pk; pk[0] = (_Float16)p.x; pk[1] = (_Float16)p.y;
            pk[2] = (_Float16)p.z; pk[3] = (_Float16)p.w;
            *(v4h*)&es_lds[e * 64 + c4] = pk;
            *(float4*)&xs_lds[e * 64 + c4] =
                *(const float4*)(xs + (size_t)src_lds[e] * 64 + c4);
        }
        if constexpr (LAYER >= 2) {
            for (int i = t; i < 16 * 24; i += 128) {   // 16 edges * 24 float4
                const int e = i / 24, r = i % 24;
                const int pl = r >> 3, c4 = (r & 7) * 4;
                *(float4*)&xv_lds[e * 96 + pl * 32 + c4] =
                    *(const float4*)(xv + pl * pvs_in + (size_t)src_lds[e] * 32 + c4);
            }
        }
        __syncthreads();

        // ---- GEMM1: h = silu(es @ W1 / 8), one 16-col tile per wave
        {
            const int n0 = wave * 16;
            v8f acc = {};
#pragma unroll
            for (int k0 = 0; k0 < 64; k0 += 32) {
                const v16h a = frag_from_lds(es_lds + row * 64 + k0, kb);
                const v16h b = frag_from_lds(W1t + (n0 + col) * 64 + k0, kb);
                acc = __builtin_amdgcn_wmma_f32_16x16x32_f16(
                    false, a, false, b, (short)0, acc, false, false);
            }
#pragma unroll
            for (int r = 0; r < 8; ++r)
                h_lds[(rb + r) * 64 + n0 + col] = (_Float16)silu_f(acc[r] * 0.125f);
        }
        __syncthreads();

        // ---- GEMM2: w = h @ W2 / 8, waves stride over NOUT/16 tiles
        for (int nt = wave; nt < NOUT / 16; nt += 4) {
            const int n0 = nt * 16;
            v8f acc = {};
#pragma unroll
            for (int k0 = 0; k0 < 64; k0 += 32) {
                const v16h a = frag_from_lds(h_lds + row * 64 + k0, kb);
                const v16h b = frag_from_lds(W2t + (n0 + col) * 64 + k0, kb);
                acc = __builtin_amdgcn_wmma_f32_16x16x32_f16(
                    false, a, false, b, (short)0, acc, false, false);
            }
#pragma unroll
            for (int r = 0; r < 8; ++r)
                w_lds[(rb + r) * NOUT + n0 + col] = acc[r] * 0.125f;
        }
        __syncthreads();

        // prefetch next tile's edge_scalars (global_prefetch_b8) while the
        // atomic-bound message phase runs
        {
            const int ntile = tile + gridDim.x;
            if (ntile < ntiles)
                __builtin_prefetch(es + (size_t)ntile * 16 * 64 + t * 8, 0, 1);
        }

        // ---- messages + scatter (scale 1/sqrt(16) = 0.25 folded in)
        const float qs = 0.25f;
        for (int i = t; i < 16 * 64; i += 128) {
            const int e = i >> 6, c = i & 63;
            const int dst = dst_lds[e];
            const float e_s = ev_lds[e * 4 + 0];
            const float xe  = xs_lds[e * 64 + c];
            atomicAdd(&outS[(size_t)dst * ldS + c], w_lds[e * NOUT + c] * xe * e_s * qs);
            if constexpr (LAYER <= 2) {
                const float mv = w_lds[e * NOUT + 64 + c] * xe * qs;
#pragma unroll
                for (int pl = 0; pl < 3; ++pl)
                    atomicAdd(&outV[pl * pvs_out + (size_t)dst * ldV + c],
                              mv * ev_lds[e * 4 + 1 + pl]);
            }
        }
        if constexpr (LAYER >= 2) {
            for (int i = t; i < 16 * 32; i += 128) {
                const int e = i >> 5, c = i & 31;
                const int dst = dst_lds[e];
                const float e_s = ev_lds[e * 4 + 0];
                const float ev0 = ev_lds[e * 4 + 1];
                const float ev1 = ev_lds[e * 4 + 2];
                const float ev2 = ev_lds[e * 4 + 3];
                const float v0 = xv_lds[e * 96 + c];
                const float v1 = xv_lds[e * 96 + 32 + c];
                const float v2 = xv_lds[e * 96 + 64 + c];
                const float dotv = v0 * ev0 + v1 * ev1 + v2 * ev2;
                if constexpr (LAYER == 2) {
                    const float w2c = w_lds[e * NOUT + 128 + c] * e_s * qs;
                    atomicAdd(&outV[0 * pvs_out + (size_t)dst * ldV + 64 + c], w2c * v0);
                    atomicAdd(&outV[1 * pvs_out + (size_t)dst * ldV + 64 + c], w2c * v1);
                    atomicAdd(&outV[2 * pvs_out + (size_t)dst * ldV + 64 + c], w2c * v2);
                    const float w3c = w_lds[e * NOUT + 160 + c];
                    atomicAdd(&outS[(size_t)dst * ldS + 64 + c], w3c * dotv * INV_SQRT3 * qs);
                } else {
                    const float w3c = w_lds[e * NOUT + 64 + c];
                    atomicAdd(&outS[(size_t)dst * ldS + 64 + c], w3c * dotv * INV_SQRT3 * qs);
                }
            }
        }
        __syncthreads();   // LDS reused next iteration
    }
}

// ---------------------------------------------------------------------------
// Layer transition: s = silu(out_s[:, :64]);  v = out_v * sigmoid(out_s[:,64:96])
// ---------------------------------------------------------------------------
__global__ void transition_kernel(const float* __restrict__ outs,  // N x 96
                                  const float* __restrict__ outv,  // 3 x N x 32
                                  float* __restrict__ s,           // N x 64
                                  float* __restrict__ v,           // 3 x N x 32
                                  int N)
{
    const int i = blockIdx.x * blockDim.x + threadIdx.x;
    if (i >= N * 96) return;
    const int n = i / 96, c = i % 96;
    const float x = outs[i];
    if (c < 64) {
        s[(size_t)n * 64 + c] = silu_f(x);
    } else {
        const int cc = c - 64;
        const float g = sigmoid_f(x);
        const size_t ps = (size_t)N * 32;
#pragma unroll
        for (int pl = 0; pl < 3; ++pl)
            v[pl * ps + (size_t)n * 32 + cc] = outv[pl * ps + (size_t)n * 32 + cc] * g;
    }
}

__global__ void zero_kernel(float* __restrict__ p, size_t n)
{
    for (size_t i = (size_t)blockIdx.x * blockDim.x + threadIdx.x; i < n;
         i += (size_t)gridDim.x * blockDim.x)
        p[i] = 0.0f;
}

// ---------------------------------------------------------------------------
extern "C" void kernel_launch(void* const* d_in, const int* in_sizes, int n_in,
                              void* d_out, int out_size, void* d_ws, size_t ws_size,
                              hipStream_t stream)
{
    const int N = N_NODES, E = N_EDGES;
    const int NT = E / 16;
    const float* nf    = (const float*)d_in[0];
    const float* na    = (const float*)d_in[1];   // (N,1) row scale
    const int*   esrc  = (const int*)d_in[2];
    const int*   edst  = (const int*)d_in[3];
    const float* eattr = (const float*)d_in[4];
    const float* escal = (const float*)d_in[5];
    const float* W[20];
    for (int i = 0; i < 20; ++i) W[i] = (const float*)d_in[6 + i];
    // 0 l1_sc_s, 1 l1_lin1_s, 2 l1_fc1, 3 l1_fc2, 4 l1_lin2_s, 5 l1_lin2_v,
    // 6 l2_sc_s, 7 l2_sc_v, 8 l2_lin1_s, 9 l2_lin1_v, 10 l2_fc1, 11 l2_fc2,
    // 12 l2_lin2_s, 13 l2_lin2_v, 14 l3_sc_s, 15 l3_lin1_s, 16 l3_lin1_v,
    // 17 l3_fc1, 18 l3_fc2, 19 l3_lin2_s

    float* wsf = (float*)d_ws;
    size_t off = 0;
    auto alloc = [&](size_t n) { float* p = wsf + off; off += n; return p; };
    float* xs   = alloc((size_t)N * 64);
    float* xv   = alloc((size_t)3 * N * 32);   // plane stride N*32
    float* sc_s = alloc((size_t)N * 96);
    float* sc_v = alloc((size_t)3 * N * 32);
    float* agS  = alloc((size_t)N * 96);
    float* agV  = alloc((size_t)3 * N * 96);   // plane stride N*96
    float* outs = alloc((size_t)N * 96);
    float* outv = alloc((size_t)3 * N * 32);
    float* scur = alloc((size_t)N * 64);
    float* vcur = alloc((size_t)3 * N * 32);
    (void)ws_size; (void)in_sizes; (void)n_in; (void)out_size;

    const size_t psV32 = (size_t)N * 32;
    const size_t psV96 = (size_t)N * 96;
    const float s8  = 0.125f;                 // 1/sqrt(64)
    const float s32 = 0.17677669529663687f;   // 1/sqrt(32)
    const float s96 = 0.10206207261596575f;   // 1/sqrt(96)

    auto gemm = [&](const float* A, const float* B, const float* resid,
                    float* C, int M, int K, int Nc, float scale) {
        dim3 grid(Nc / 16, (M + 63) / 64);
        if (resid)
            gemm_wmma_f16<0, true ><<<grid, 128, 0, stream>>>(A, na, B, resid, C, M, K, Nc, scale);
        else
            gemm_wmma_f16<0, false><<<grid, 128, 0, stream>>>(A, na, B, resid, C, M, K, Nc, scale);
    };
    auto zero_ag = [&]() {
        zero_kernel<<<2048, 256, 0, stream>>>(agS, (size_t)N * 96 + (size_t)3 * N * 96);
    };

    // ================= Layer 1 =================
    gemm(nf, W[0], nullptr, sc_s, N, 64, 96, s8);
    gemm(nf, W[1], nullptr, xs,   N, 64, 64, s8);
    zero_ag();
    edge_fused<1><<<EDGE_BLOCKS, 128, 0, stream>>>(escal, eattr, esrc, edst,
        W[2], W[3], xs, nullptr, 0, agS, 64, agV, 64, psV96, NT);
    gemm(agS, W[4], sc_s, outs, N, 64, 96, s8);
    for (int pl = 0; pl < 3; ++pl)
        gemm(agV + pl * psV96, W[5], nullptr, outv + pl * psV32, N, 64, 32, s8);
    transition_kernel<<<(N * 96 + 255) / 256, 256, 0, stream>>>(outs, outv, scur, vcur, N);

    // ================= Layer 2 =================
    gemm(scur, W[6], nullptr, sc_s, N, 64, 96, s8);
    for (int pl = 0; pl < 3; ++pl)
        gemm(vcur + pl * psV32, W[7], nullptr, sc_v + pl * psV32, N, 32, 32, s32);
    gemm(scur, W[8], nullptr, xs, N, 64, 64, s8);
    for (int pl = 0; pl < 3; ++pl)
        gemm(vcur + pl * psV32, W[9], nullptr, xv + pl * psV32, N, 32, 32, s32);
    zero_ag();
    edge_fused<2><<<EDGE_BLOCKS, 128, 0, stream>>>(escal, eattr, esrc, edst,
        W[10], W[11], xs, xv, psV32, agS, 96, agV, 96, psV96, NT);
    gemm(agS, W[12], sc_s, outs, N, 96, 96, s96);
    for (int pl = 0; pl < 3; ++pl)
        gemm(agV + pl * psV96, W[13], sc_v + pl * psV32, outv + pl * psV32, N, 96, 32, s96);
    transition_kernel<<<(N * 96 + 255) / 256, 256, 0, stream>>>(outs, outv, scur, vcur, N);

    // ================= Layer 3 =================
    gemm(scur, W[14], nullptr, sc_s, N, 64, 64, s8);
    gemm(scur, W[15], nullptr, xs, N, 64, 64, s8);
    for (int pl = 0; pl < 3; ++pl)
        gemm(vcur + pl * psV32, W[16], nullptr, xv + pl * psV32, N, 32, 32, s32);
    zero_ag();
    edge_fused<3><<<EDGE_BLOCKS, 128, 0, stream>>>(escal, eattr, esrc, edst,
        W[17], W[18], xs, xv, psV32, agS, 96, nullptr, 0, 0, NT);
    gemm(agS, W[19], sc_s, (float*)d_out, N, 96, 64, s96);
}
